// ScaledDotProductAttention_58093727646286
// MI455X (gfx1250) — matrix operand order repaired and checked by reading, not compiled, not run on hardware
//
#include <hip/hip_runtime.h>

// ---------------------------------------------------------------------------
// Flash-attention for gfx1250 (CDNA5, wave32) using v_wmma_f32_16x16x32_f16.
// B=32, LQ=LK=2048, D=128, fp32 in/out, bool mask (True = masked out).
// Reference quirk preserved: score MULTIPLIED by sqrt(D); fully-masked rows -> 0.
// Double-buffered K/V tiles in LDS. A scheduling fence between the LDS
// B-operand loads and the WMMA chain stops the scheduler from sinking each
// ds_load to its consumer (which serialized every WMMA behind an LDS round
// trip in the previous build).
// ---------------------------------------------------------------------------

typedef _Float16 h16 __attribute__((ext_vector_type(16)));
typedef _Float16 h8  __attribute__((ext_vector_type(8)));
typedef float    f8  __attribute__((ext_vector_type(8)));

#define BATCH 32
#define SEQ_Q 2048
#define SEQ_K 2048
#define DIM   128
#define BM    128          // q rows per block (8 waves x 16 rows)
#define BN    32           // k columns per main-loop iteration
#define NIT   (SEQ_K / BN) // 64 iterations
#define KS_STRIDE 136      // halves per K-tile row (128 + 8 pad -> conflict-free b128)
#define VT_STRIDE 40       // halves per Vt row (32 + 8 pad, 16B-aligned)
#define PS_STRIDE 40       // halves per P-scratch row

#define KS_HALFS (32 * KS_STRIDE)          // 4352
#define VT_HALFS (128 * VT_STRIDE)         // 5120
#define BUF_HALFS (KS_HALFS + VT_HALFS)    // 9472 per K/V buffer
#define PS_HALFS (8 * 16 * PS_STRIDE)      // 5120

// Zero-cost scheduling fence: memory ops may not cross it in either direction.
#define SCHED_FENCE() asm volatile("" ::: "memory")

__device__ __forceinline__ h16 cat8(h8 lo, h8 hi) {
  return __builtin_shufflevector(lo, hi, 0,1,2,3,4,5,6,7,8,9,10,11,12,13,14,15);
}

// Issue the global float4 loads for one 32x128 K tile + V tile (per-thread slice).
__device__ __forceinline__ void load_tile(const float* __restrict__ kb,
                                          const float* __restrict__ vb,
                                          int kk, int tid,
                                          float4 kf[4], float4 vf[4]) {
#pragma unroll
  for (int g = 0; g < 4; ++g) {
    const int idx = (tid + 256 * g) * 4;   // 4096 elements, 16 per thread
    const int r = idx >> 7;
    const int d = idx & 127;
    kf[g] = *(const float4*)(kb + (size_t)(kk + r) * DIM + d);
    vf[g] = *(const float4*)(vb + (size_t)(kk + r) * DIM + d);
  }
}

// Convert the staged registers to f16 and store into the LDS tile buffers.
__device__ __forceinline__ void stage_tile(_Float16* __restrict__ Ks,
                                           _Float16* __restrict__ Vt,
                                           int tid,
                                           const float4 kf[4], const float4 vf[4]) {
#pragma unroll
  for (int g = 0; g < 4; ++g) {
    const int idx = (tid + 256 * g) * 4;
    const int r = idx >> 7;
    const int d = idx & 127;
    Ks[r * KS_STRIDE + d + 0] = (_Float16)kf[g].x;
    Ks[r * KS_STRIDE + d + 1] = (_Float16)kf[g].y;
    Ks[r * KS_STRIDE + d + 2] = (_Float16)kf[g].z;
    Ks[r * KS_STRIDE + d + 3] = (_Float16)kf[g].w;
    Vt[(d + 0) * VT_STRIDE + r] = (_Float16)vf[g].x;
    Vt[(d + 1) * VT_STRIDE + r] = (_Float16)vf[g].y;
    Vt[(d + 2) * VT_STRIDE + r] = (_Float16)vf[g].z;
    Vt[(d + 3) * VT_STRIDE + r] = (_Float16)vf[g].w;
  }
}

__global__ __launch_bounds__(256)
void ScaledDotProductAttention_58093727646286_kernel(
    const float* __restrict__ Q, const float* __restrict__ K,
    const float* __restrict__ V, const unsigned char* __restrict__ M,
    float* __restrict__ O) {
  __shared__ _Float16 lds[2 * BUF_HALFS + PS_HALFS];
  _Float16* Ps = lds + 2 * BUF_HALFS;       // [8 waves][16][40] P scratch

  const int tid   = threadIdx.x;
  const int wave  = tid >> 5;
  const int lane  = tid & 31;
  const int row   = lane & 15;   // N index inside a 16-wide tile / M index for A
  const int hhalf = lane >> 4;   // which 16-lane half of the wave

  const int blk = blockIdx.x;
  const int b   = blk >> 4;                  // SEQ_Q / BM == 16 blocks per batch
  const int q0  = (blk & 15) * BM;

  const float NEG_INF = -__builtin_inff();
  const float SCALE   = 11.313708498984760f; // sqrt(128) -- reference multiplies!

  const float* qb = Q + (size_t)b * SEQ_Q * DIM;
  const float* kb = K + (size_t)b * SEQ_K * DIM;
  const float* vb = V + (size_t)b * SEQ_K * DIM;
  const unsigned char* mb = M + (size_t)b * SEQ_Q * SEQ_K;

  // ---- Load this wave's Q (16 rows x 128) into WMMA A-layout registers ----
  // A layout (f16 16x32): lane<16 : M=lane, K = {0..7, 16..23} (+32c)
  //                       lane>=16: M=lane-16, K = {8..15, 24..31} (+32c)
  const int qrow = q0 + wave * 16 + row;
  const float* qp = qb + (size_t)qrow * DIM;
  h16 qa[4];
#pragma unroll
  for (int c = 0; c < 4; ++c) {
    const int d0 = 32 * c + 8 * hhalf;
    float4 f0 = *(const float4*)(qp + d0);
    float4 f1 = *(const float4*)(qp + d0 + 4);
    float4 f2 = *(const float4*)(qp + d0 + 16);
    float4 f3 = *(const float4*)(qp + d0 + 20);
    qa[c][0]  = (_Float16)f0.x; qa[c][1]  = (_Float16)f0.y;
    qa[c][2]  = (_Float16)f0.z; qa[c][3]  = (_Float16)f0.w;
    qa[c][4]  = (_Float16)f1.x; qa[c][5]  = (_Float16)f1.y;
    qa[c][6]  = (_Float16)f1.z; qa[c][7]  = (_Float16)f1.w;
    qa[c][8]  = (_Float16)f2.x; qa[c][9]  = (_Float16)f2.y;
    qa[c][10] = (_Float16)f2.z; qa[c][11] = (_Float16)f2.w;
    qa[c][12] = (_Float16)f3.x; qa[c][13] = (_Float16)f3.y;
    qa[c][14] = (_Float16)f3.z; qa[c][15] = (_Float16)f3.w;
  }

  // O accumulators: 16x128 f32 as 8 C-tiles (d-chunks of 16). C layout:
  // VGPR j holds row M=j (lanes 0-15) / M=j+8 (lanes 16-31), col N=lane%16.
  f8 acc[8] = {};
  float mrow[8], lrow[8];
#pragma unroll
  for (int j = 0; j < 8; ++j) { mrow[j] = NEG_INF; lrow[j] = 0.f; }

  // Per-lane mask row pointers (row of element j is j + 8*hhalf)
  const unsigned char* mrp[8];
#pragma unroll
  for (int j = 0; j < 8; ++j)
    mrp[j] = mb + (size_t)(q0 + wave * 16 + j + 8 * hhalf) * SEQ_K;

  _Float16* pw = Ps + wave * (16 * PS_STRIDE);

  // ---- Prologue: stage tile 0 into buffer 0 ----
  float4 kf[4], vf[4];
  load_tile(kb, vb, 0, tid, kf, vf);
  stage_tile(lds, lds + KS_HALFS, tid, kf, vf);
  __syncthreads();

  for (int it = 0; it < NIT; ++it) {
    const int kk  = it * BN;
    const int cur = it & 1;
    _Float16* Ks = lds + cur * BUF_HALFS;
    _Float16* Vt = Ks + KS_HALFS;
    const bool has_next = (it + 1) < NIT;

    // Issue next tile's global loads early; results consumed at the end of
    // this iteration (LOADcnt keeps them in flight across the WMMA work).
    if (has_next) load_tile(kb, vb, kk + BN, tid, kf, vf);

    // ---- S = Q * K^T : preload all 8 B operands, then 8 WMMAs back-to-back.
    // B layout (f16, K=32): lane L holds col N=L%16, K = {0..7,16..23} (+8 for L>=16).
    h16 bk[8];
#pragma unroll
    for (int c = 0; c < 4; ++c) {
      const _Float16* kr0 = Ks + row * KS_STRIDE + 32 * c + 8 * hhalf;
      const _Float16* kr1 = kr0 + 16 * KS_STRIDE;
      bk[2 * c + 0] = cat8(*(const h8*)kr0, *(const h8*)(kr0 + 16));
      bk[2 * c + 1] = cat8(*(const h8*)kr1, *(const h8*)(kr1 + 16));
    }
    SCHED_FENCE();  // keep all 16 ds_load_b128 above the WMMA chain
    f8 s0 = {}, s1 = {};
#pragma unroll
    for (int c = 0; c < 4; ++c) {
      s0 = __builtin_amdgcn_wmma_f32_16x16x32_f16(false, qa[c], false, bk[2 * c + 0],
                                                  (short)0, s0, false, false);
      s1 = __builtin_amdgcn_wmma_f32_16x16x32_f16(false, qa[c], false, bk[2 * c + 1],
                                                  (short)0, s1, false, false);
    }

    // ---- scale, mask, online softmax (per row = per (j, half)) ----
#pragma unroll
    for (int j = 0; j < 8; ++j) {
      float sv0 = s0[j] * SCALE;
      float sv1 = s1[j] * SCALE;
      const unsigned char m0 = mrp[j][kk + row];
      const unsigned char m1 = mrp[j][kk + 16 + row];
      sv0 = m0 ? NEG_INF : sv0;
      sv1 = m1 ? NEG_INF : sv1;

      float mx = fmaxf(sv0, sv1);                 // row max over 16 lanes
#pragma unroll
      for (int o = 1; o < 16; o <<= 1) mx = fmaxf(mx, __shfl_xor(mx, o, 32));

      const float mold = mrow[j];
      const float mnew = fmaxf(mold, mx);
      const float alpha = (mold == NEG_INF) ? 0.f : __expf(mold - mnew);
      float p0 = (sv0 == NEG_INF) ? 0.f : __expf(sv0 - mnew);
      float p1 = (sv1 == NEG_INF) ? 0.f : __expf(sv1 - mnew);

      float ps = p0 + p1;                          // row sum over 16 lanes
#pragma unroll
      for (int o = 1; o < 16; o <<= 1) ps += __shfl_xor(ps, o, 32);

      lrow[j] = lrow[j] * alpha + ps;
      mrow[j] = mnew;
#pragma unroll
      for (int dch = 0; dch < 8; ++dch) acc[dch][j] *= alpha;

      // Stage P (C layout) into per-wave LDS scratch for A-layout reload.
      const int prow = j + 8 * hhalf;
      pw[prow * PS_STRIDE + row]      = (_Float16)p0;
      pw[prow * PS_STRIDE + 16 + row] = (_Float16)p1;
    }

    // Intra-wave LDS handoff (cross-lane): wait for DS stores, block reorder.
    asm volatile("s_wait_dscnt 0x0" ::: "memory");

    // ---- Reload P in A layout; preload V B operands; 8 WMMAs back-to-back.
    const _Float16* pr = pw + row * PS_STRIDE;
    h16 pa = cat8(*(const h8*)(pr + 8 * hhalf),
                  *(const h8*)(pr + 16 + 8 * hhalf));
    h16 bv[8];
#pragma unroll
    for (int dch = 0; dch < 8; ++dch) {
      const _Float16* vr = Vt + (16 * dch + row) * VT_STRIDE + 8 * hhalf;
      bv[dch] = cat8(*(const h8*)vr, *(const h8*)(vr + 16));
    }
    SCHED_FENCE();  // keep the 16 V-operand ds_load_b128 above the WMMA chain
#pragma unroll
    for (int dch = 0; dch < 8; ++dch) {
      acc[dch] = __builtin_amdgcn_wmma_f32_16x16x32_f16(false, pa, false, bv[dch],
                                                        (short)0, acc[dch],
                                                        false, false);
    }

    // ---- Stage next tile into the other buffer; one barrier per iteration.
    if (has_next) {
      _Float16* KsN = lds + (cur ^ 1) * BUF_HALFS;
      stage_tile(KsN, KsN + KS_HALFS, tid, kf, vf);
    }
    __syncthreads();
  }

  // ---- Epilogue: normalize (l==0 -> 0, matching the NaN->0 reference) ----
#pragma unroll
  for (int j = 0; j < 8; ++j) {
    const float inv = (lrow[j] > 0.f) ? (1.f / lrow[j]) : 0.f;
    const int gq = q0 + wave * 16 + j + 8 * hhalf;
    float* op = O + ((size_t)b * SEQ_Q + gq) * DIM;
#pragma unroll
    for (int dch = 0; dch < 8; ++dch) op[16 * dch + row] = acc[dch][j] * inv;
  }
}

extern "C" void kernel_launch(void* const* d_in, const int* in_sizes, int n_in,
                              void* d_out, int out_size, void* d_ws, size_t ws_size,
                              hipStream_t stream) {
  (void)in_sizes; (void)n_in; (void)out_size; (void)d_ws; (void)ws_size;
  const float* q = (const float*)d_in[0];
  const float* k = (const float*)d_in[1];
  const float* v = (const float*)d_in[2];
  const unsigned char* mask = (const unsigned char*)d_in[3];
  float* out = (float*)d_out;

  dim3 grid(BATCH * (SEQ_Q / BM));  // 512 blocks
  dim3 block(256);                  // 8 waves, one 16-row q-stripe each
  ScaledDotProductAttention_58093727646286_kernel<<<grid, block, 0, stream>>>(
      q, k, v, mask, out);
}
